// DeltaNet_61246233641414
// MI455X (gfx1250) — compile-verified, matrix operand order checked by reading, not run on hardware
//
#include <hip/hip_runtime.h>
#include <hip/hip_bf16.h>

typedef __attribute__((ext_vector_type(16))) _Float16 v16h;
typedef __attribute__((ext_vector_type(8)))  float    v8f;

#define B_   2
#define S_   2048
#define D_   1024
#define H_   16
#define DH_  64
#define C_   2
#define MTOK (B_ * S_)   // 4096

// ---------------------------------------------------------------------------
// CDNA5 async global->LDS copy (ASYNCcnt path), with sync fallback
// ---------------------------------------------------------------------------
#if defined(__has_builtin)
#if __has_builtin(__builtin_amdgcn_global_load_async_to_lds_b128)
#define HAVE_ASYNC_LDS 1
#endif
#endif

#if defined(HAVE_ASYNC_LDS)
// builtin signature: void(v4i addrspace(1)*, v4i addrspace(3)*, imm offset, imm cpol)
typedef int v4i_vs __attribute__((vector_size(4 * sizeof(int))));
typedef __attribute__((address_space(1))) v4i_vs* g_v4i_p;
typedef __attribute__((address_space(3))) v4i_vs* l_v4i_p;
__device__ __forceinline__ void async_copy_b128(const _Float16* g, _Float16* l) {
    __builtin_amdgcn_global_load_async_to_lds_b128(
        (g_v4i_p)(void*)const_cast<_Float16*>(g), (l_v4i_p)(void*)l, 0, 0);
}
#if __has_builtin(__builtin_amdgcn_s_wait_asynccnt)
#define WAIT_ASYNC(n) __builtin_amdgcn_s_wait_asynccnt(n)
#else
#define WAIT_ASYNC(n) asm volatile("s_wait_asynccnt %0" :: "n"(n) : "memory")
#endif
#endif

// ---------------------------------------------------------------------------
// f32 -> f16 conversion
// ---------------------------------------------------------------------------
__global__ __launch_bounds__(256) void f32_to_f16_kernel(const float* __restrict__ src,
                                                         _Float16* __restrict__ dst, int n) {
    int i = blockIdx.x * 256 + threadIdx.x;
    if (i < n) dst[i] = (_Float16)src[i];
}

// W is (K x N) row-major f32; Wt becomes (N x K) row-major f16
__global__ __launch_bounds__(256) void transpose_to_f16_kernel(const float* __restrict__ W,
                                                               _Float16* __restrict__ Wt,
                                                               int K, int N) {
    int i = blockIdx.x * 256 + threadIdx.x;
    if (i < K * N) {
        int k = i / N, n = i % N;
        Wt[(size_t)n * K + k] = (_Float16)W[i];
    }
}

// ---------------------------------------------------------------------------
// WMMA GEMM:  C[M,N] = A[M,K] (f16) * Bt[N,K]^T (f16)  (+ resid + bias)
// block = 256 threads (8 waves), tile = 128x128, K-step = 32,
// double-buffered async global->LDS staging.
// ---------------------------------------------------------------------------
__device__ inline v16h load_frag_k32(const _Float16* tile_base, int row16, int half) {
    // tile rows have stride 32 f16; A-fragment layout for v_wmma_f32_16x16x32_f16:
    // lanes 0-15: K = {0..7, 16..23}; lanes 16-31: K = {8..15, 24..31}
    const _Float16* p = tile_base + row16 * 32 + half * 8;
    union { v16h h; uint4 u[2]; } f;
    f.u[0] = *(const uint4*)(p);
    f.u[1] = *(const uint4*)(p + 16);
    return f.h;
}

__global__ __launch_bounds__(256) void gemm_f16_wmma_kernel(
    const _Float16* __restrict__ A,   // M x K row-major
    const _Float16* __restrict__ Bt,  // N x K row-major (B pre-transposed)
    float* __restrict__ Cout,         // M x N
    const float* __restrict__ resid,  // optional M x N, may be null
    const float* __restrict__ bias,   // optional N, may be null
    int M, int N, int K)
{
#if defined(HAVE_ASYNC_LDS)
    __shared__ _Float16 As[2][128 * 32];
    __shared__ _Float16 Bs[2][128 * 32];
#else
    __shared__ _Float16 As[1][128 * 32];
    __shared__ _Float16 Bs[1][128 * 32];
#endif

    const int t     = threadIdx.x;
    const int m0    = blockIdx.y * 128;
    const int n0    = blockIdx.x * 128;
    const int wave  = t >> 5;
    const int lane  = t & 31;
    const int half  = (lane >> 4) & 1;
    const int l16   = lane & 15;
    const int waveM = wave >> 2;   // 0..1  (64-row slab)
    const int waveN = wave & 3;    // 0..3  (32-col slab)

    // per-thread staging coordinates: 2 chunks of 8 f16 (16B) per matrix
    const int r0  = t >> 2;              // chunk 0 row
    const int cg0 = t & 3;               // chunk 0 col-group
    const int r1  = (t + 256) >> 2;      // chunk 1 row
    const int cg1 = t & 3;

    v8f acc[4][2] = {};
    const int nIter = K / 32;

#if defined(HAVE_ASYNC_LDS)
    // prologue: stage tile 0 into buffer 0 (4 async-load instructions / wave)
    {
        const int k0 = 0;
        async_copy_b128(&A [(size_t)(m0 + r0) * K + k0 + cg0 * 8], &As[0][r0 * 32 + cg0 * 8]);
        async_copy_b128(&Bt[(size_t)(n0 + r0) * K + k0 + cg0 * 8], &Bs[0][r0 * 32 + cg0 * 8]);
        async_copy_b128(&A [(size_t)(m0 + r1) * K + k0 + cg1 * 8], &As[0][r1 * 32 + cg1 * 8]);
        async_copy_b128(&Bt[(size_t)(n0 + r1) * K + k0 + cg1 * 8], &Bs[0][r1 * 32 + cg1 * 8]);
    }
    for (int it = 0; it < nIter; ++it) {
        const int cur = it & 1;
        if (it + 1 < nIter) {
            const int k0 = (it + 1) * 32;
            const int nxt = cur ^ 1;
            async_copy_b128(&A [(size_t)(m0 + r0) * K + k0 + cg0 * 8], &As[nxt][r0 * 32 + cg0 * 8]);
            async_copy_b128(&Bt[(size_t)(n0 + r0) * K + k0 + cg0 * 8], &Bs[nxt][r0 * 32 + cg0 * 8]);
            async_copy_b128(&A [(size_t)(m0 + r1) * K + k0 + cg1 * 8], &As[nxt][r1 * 32 + cg1 * 8]);
            async_copy_b128(&Bt[(size_t)(n0 + r1) * K + k0 + cg1 * 8], &Bs[nxt][r1 * 32 + cg1 * 8]);
            WAIT_ASYNC(4);   // drain previous batch only; next batch stays in flight
        } else {
            WAIT_ASYNC(0);
        }
        __syncthreads();

        v16h af[4], bf[2];
        #pragma unroll
        for (int mt = 0; mt < 4; ++mt)
            af[mt] = load_frag_k32(&As[cur][(waveM * 64 + mt * 16) * 32], l16, half);
        #pragma unroll
        for (int nt = 0; nt < 2; ++nt)
            bf[nt] = load_frag_k32(&Bs[cur][(waveN * 32 + nt * 16) * 32], l16, half);

        #pragma unroll
        for (int mt = 0; mt < 4; ++mt)
            #pragma unroll
            for (int nt = 0; nt < 2; ++nt)
                acc[mt][nt] = __builtin_amdgcn_wmma_f32_16x16x32_f16(
                    false, af[mt], false, bf[nt], (short)0, acc[mt][nt], false, false);
        __syncthreads();   // all reads of buf[cur] done before it is restaged
    }
#else
    for (int it = 0; it < nIter; ++it) {
        const int k0 = it * 32;
        *(uint4*)&As[0][r0 * 32 + cg0 * 8] = *(const uint4*)&A [(size_t)(m0 + r0) * K + k0 + cg0 * 8];
        *(uint4*)&Bs[0][r0 * 32 + cg0 * 8] = *(const uint4*)&Bt[(size_t)(n0 + r0) * K + k0 + cg0 * 8];
        *(uint4*)&As[0][r1 * 32 + cg1 * 8] = *(const uint4*)&A [(size_t)(m0 + r1) * K + k0 + cg1 * 8];
        *(uint4*)&Bs[0][r1 * 32 + cg1 * 8] = *(const uint4*)&Bt[(size_t)(n0 + r1) * K + k0 + cg1 * 8];
        if (k0 + 32 < K) {
            __builtin_prefetch(&A [(size_t)(m0 + (t >> 1)) * K + k0 + 32], 0, 0);
            __builtin_prefetch(&Bt[(size_t)(n0 + (t >> 1)) * K + k0 + 32], 0, 0);
        }
        __syncthreads();
        v16h af[4], bf[2];
        #pragma unroll
        for (int mt = 0; mt < 4; ++mt)
            af[mt] = load_frag_k32(&As[0][(waveM * 64 + mt * 16) * 32], l16, half);
        #pragma unroll
        for (int nt = 0; nt < 2; ++nt)
            bf[nt] = load_frag_k32(&Bs[0][(waveN * 32 + nt * 16) * 32], l16, half);
        #pragma unroll
        for (int mt = 0; mt < 4; ++mt)
            #pragma unroll
            for (int nt = 0; nt < 2; ++nt)
                acc[mt][nt] = __builtin_amdgcn_wmma_f32_16x16x32_f16(
                    false, af[mt], false, bf[nt], (short)0, acc[mt][nt], false, false);
        __syncthreads();
    }
#endif

    // C/D layout: VGPR i -> M = i + 8*half, N = l16
    #pragma unroll
    for (int mt = 0; mt < 4; ++mt) {
        #pragma unroll
        for (int nt = 0; nt < 2; ++nt) {
            #pragma unroll
            for (int i = 0; i < 8; ++i) {
                int m = m0 + waveM * 64 + mt * 16 + 8 * half + i;
                int n = n0 + waveN * 32 + nt * 16 + l16;
                float v = acc[mt][nt][i];
                if (resid) v += resid[(size_t)m * N + n];
                if (bias)  v += bias[n];
                Cout[(size_t)m * N + n] = v;
            }
        }
    }
}

// ---------------------------------------------------------------------------
// beta = clip(sigmoid(x @ Wbeta + bbeta), 1e-3, 0.999)   (4096 x 32)
// ---------------------------------------------------------------------------
__global__ __launch_bounds__(128) void beta_kernel(const float* __restrict__ x,
                                                   const float* __restrict__ Wb,
                                                   const float* __restrict__ bb,
                                                   float* __restrict__ beta) {
    __shared__ float pb[128];
    int m = blockIdx.x;
    int t = threadIdx.x;
    int c = t & 31, seg = t >> 5;
    const float* xr = x + (size_t)m * D_;
    float s = 0.f;
    int kb = seg * 256;
    for (int k = kb; k < kb + 256; ++k)
        s += xr[k] * Wb[(size_t)k * 32 + c];
    pb[t] = s;
    __syncthreads();
    if (t < 32) {
        float tot = pb[t] + pb[t + 32] + pb[t + 64] + pb[t + 96];
        float logit = tot + bb[t];
        float sg = 1.f / (1.f + __expf(-logit));
        sg = fminf(fmaxf(sg, 0.001f), 0.999f);
        beta[(size_t)m * 32 + t] = sg;
    }
}

// ---------------------------------------------------------------------------
// RoPE + phi (elu+1), in place, per head half-pair (i, i+32)
// ---------------------------------------------------------------------------
__global__ __launch_bounds__(256) void rope_phi_kernel(float* __restrict__ u, int total) {
    int idx = blockIdx.x * 256 + threadIdx.x;
    if (idx >= total) return;
    int i   = idx & 31;        // rotation-frequency index
    int rem = idx >> 5;
    int hh  = rem & 15;        // head
    int m   = rem >> 4;        // token 0..4095
    int s   = m & (S_ - 1);    // position within sequence
    float f   = __expf(-(float)(2 * i) * 9.210340371976184f / 64.f); // ln(10000)
    float ang = (float)s * f;
    float cs = __cosf(ang), sn = __sinf(ang);
    size_t base = (size_t)m * D_ + hh * DH_;
    float u1 = u[base + i], u2 = u[base + 32 + i];
    float r1 = u1 * cs - u2 * sn;
    float r2 = u1 * sn + u2 * cs;
    u[base + i]      = r1 > 0.f ? r1 + 1.f : __expf(r1);  // elu(r)+1
    u[base + 32 + i] = r2 > 0.f ? r2 + 1.f : __expf(r2);
}

// ---------------------------------------------------------------------------
// Sequential DeltaNet scan: one block per (b,h). State M[C][64][64] lives in
// registers (32 f32/thread over 256 threads), Z[C][64] in threads 0..127.
// Writes y as f16 (feeds the output WMMA GEMM directly).
// ---------------------------------------------------------------------------
__global__ __launch_bounds__(256) void scan_kernel(const float* __restrict__ q,
                                                   const float* __restrict__ k,
                                                   const float* __restrict__ v,
                                                   const float* __restrict__ beta,
                                                   _Float16* __restrict__ y) {
    __shared__ float sq[DH_], sk[DH_], sv[DH_], sb[C_];
    __shared__ float partial[4][DH_];
    __shared__ float denb[4];

    int bh = blockIdx.x;
    int b = bh / H_, h = bh % H_;
    int t    = threadIdx.x;
    int lane = t & 31, wv = t >> 5;
    int e  = t & 63;           // output dim owned by this thread
    int g  = t >> 6;           // group 0..3
    int c  = g >> 1;           // decay channel
    int d0 = (g & 1) * 32;     // k-dim slab

    float mreg[32];
    #pragma unroll
    for (int i = 0; i < 32; ++i) mreg[i] = 0.f;
    float zreg = 0.f;
    int zc = (t >> 6) & 1;     // for t<128: Z channel
    int zd = t & 63;

    const float* qp = q + (size_t)b * S_ * D_ + h * DH_;
    const float* kp = k + (size_t)b * S_ * D_ + h * DH_;
    const float* vp = v + (size_t)b * S_ * D_ + h * DH_;
    const float* bp = beta + (size_t)b * S_ * (H_ * C_) + h * C_;
    _Float16*    yp = y + (size_t)b * S_ * D_ + h * DH_;

    for (int ts = 0; ts < S_; ++ts) {
        size_t ro = (size_t)ts * D_;
        if (t < 64)        sq[t]       = qp[ro + t];
        else if (t < 128)  sk[t - 64]  = kp[ro + (t - 64)];
        else if (t < 192)  sv[t - 128] = vp[ro + (t - 128)];
        else if (t < 194)  sb[t - 192] = bp[(size_t)ts * (H_ * C_) + (t - 192)];
        __syncthreads();

        // M update + per-thread partial of num[e]
        float bc = sb[c];
        float ve = sv[e];
        float np = 0.f;
        #pragma unroll
        for (int i = 0; i < 32; ++i) {
            mreg[i] = mreg[i] * bc + sk[d0 + i] * ve;
            np += sq[d0 + i] * mreg[i];
        }
        partial[g][e] = np;

        // Z update + denominator (wave32 shuffle reduction, waves 0..3)
        if (t < 128) {
            zreg = zreg * sb[zc] + sk[zd];
            float dt = sq[zd] * zreg;
            #pragma unroll
            for (int off = 16; off > 0; off >>= 1) dt += __shfl_xor(dt, off);
            if (lane == 0) denb[wv] = dt;
        }
        __syncthreads();

        if (t < 64) {
            float num = partial[0][t] + partial[1][t] + partial[2][t] + partial[3][t];
            float den = denb[0] + denb[1] + denb[2] + denb[3] + 1e-6f;
            yp[ro + t] = (_Float16)(num / den);
        }
        __syncthreads();
    }
}

// ---------------------------------------------------------------------------
// LayerNorm over rows of h (4096 x 1024), f32 out
// ---------------------------------------------------------------------------
__global__ __launch_bounds__(256) void ln_kernel(const float* __restrict__ hbuf,
                                                 const float* __restrict__ gamma,
                                                 const float* __restrict__ lbeta,
                                                 float* __restrict__ out) {
    __shared__ float rs[256], rq[256];
    __shared__ float smu, srstd;
    int m = blockIdx.x, t = threadIdx.x;
    const float* hr = hbuf + (size_t)m * D_;
    float s = 0.f, q2 = 0.f;
    float vals[4];
    #pragma unroll
    for (int i = 0; i < 4; ++i) {
        float v = hr[t + i * 256];
        vals[i] = v; s += v; q2 += v * v;
    }
    rs[t] = s; rq[t] = q2;
    __syncthreads();
    for (int off = 128; off > 0; off >>= 1) {
        if (t < off) { rs[t] += rs[t + off]; rq[t] += rq[t + off]; }
        __syncthreads();
    }
    if (t == 0) {
        float mu  = rs[0] * (1.f / D_);
        float var = rq[0] * (1.f / D_) - mu * mu;
        smu = mu; srstd = rsqrtf(var + 1e-5f);
    }
    __syncthreads();
    float* orow = out + (size_t)m * D_;
    #pragma unroll
    for (int i = 0; i < 4; ++i) {
        int n = t + i * 256;
        orow[n] = gamma[n] * (vals[i] - smu) * srstd + lbeta[n];
    }
}

// ---------------------------------------------------------------------------
// launch
// ---------------------------------------------------------------------------
extern "C" void kernel_launch(void* const* d_in, const int* in_sizes, int n_in,
                              void* d_out, int out_size, void* d_ws, size_t ws_size,
                              hipStream_t stream) {
    const float* x      = (const float*)d_in[0];
    const float* Wq     = (const float*)d_in[1];
    const float* Wk     = (const float*)d_in[2];
    const float* Wv     = (const float*)d_in[3];
    const float* Wbeta  = (const float*)d_in[4];
    const float* bbeta  = (const float*)d_in[5];
    const float* Wo     = (const float*)d_in[6];
    const float* bo     = (const float*)d_in[7];
    const float* lng    = (const float*)d_in[8];
    const float* lnb    = (const float*)d_in[9];
    float* out = (float*)d_out;

    char* ws = (char*)d_ws;
    size_t off = 0;
    auto alloc = [&](size_t bytes) { char* p = ws + off; off += (bytes + 255) & ~(size_t)255; return p; };

    _Float16* xh   = (_Float16*)alloc((size_t)MTOK * D_ * 2);   // 8 MB
    _Float16* Wqt  = (_Float16*)alloc((size_t)D_ * D_ * 2);     // 2 MB each
    _Float16* Wkt  = (_Float16*)alloc((size_t)D_ * D_ * 2);
    _Float16* Wvt  = (_Float16*)alloc((size_t)D_ * D_ * 2);
    _Float16* Wot  = (_Float16*)alloc((size_t)D_ * D_ * 2);
    float*    qbuf = (float*)alloc((size_t)MTOK * D_ * 4);      // 16 MB each
    float*    kbuf = (float*)alloc((size_t)MTOK * D_ * 4);
    float*    vbuf = (float*)alloc((size_t)MTOK * D_ * 4);
    float*    bbuf = (float*)alloc((size_t)MTOK * 32 * 4);      // 0.5 MB
    _Float16* ybuf = (_Float16*)alloc((size_t)MTOK * D_ * 2);   // 8 MB
    float*    hbuf = (float*)alloc((size_t)MTOK * D_ * 4);      // 16 MB
    (void)ws_size; (void)in_sizes; (void)n_in; (void)out_size;

    const int nX = MTOK * D_;  // 4194304
    f32_to_f16_kernel<<<(nX + 255) / 256, 256, 0, stream>>>(x, xh, nX);

    const int nW = D_ * D_;
    transpose_to_f16_kernel<<<(nW + 255) / 256, 256, 0, stream>>>(Wq, Wqt, D_, D_);
    transpose_to_f16_kernel<<<(nW + 255) / 256, 256, 0, stream>>>(Wk, Wkt, D_, D_);
    transpose_to_f16_kernel<<<(nW + 255) / 256, 256, 0, stream>>>(Wv, Wvt, D_, D_);
    transpose_to_f16_kernel<<<(nW + 255) / 256, 256, 0, stream>>>(Wo, Wot, D_, D_);

    dim3 ggrid(D_ / 128, MTOK / 128);   // (8, 32)
    gemm_f16_wmma_kernel<<<ggrid, 256, 0, stream>>>(xh, Wqt, qbuf, nullptr, nullptr, MTOK, D_, D_);
    gemm_f16_wmma_kernel<<<ggrid, 256, 0, stream>>>(xh, Wkt, kbuf, nullptr, nullptr, MTOK, D_, D_);
    gemm_f16_wmma_kernel<<<ggrid, 256, 0, stream>>>(xh, Wvt, vbuf, nullptr, nullptr, MTOK, D_, D_);

    beta_kernel<<<MTOK, 128, 0, stream>>>(x, Wbeta, bbeta, bbuf);

    const int nPairs = MTOK * H_ * 32;  // 2097152
    rope_phi_kernel<<<nPairs / 256, 256, 0, stream>>>(qbuf, nPairs);
    rope_phi_kernel<<<nPairs / 256, 256, 0, stream>>>(kbuf, nPairs);

    scan_kernel<<<B_ * H_, 256, 0, stream>>>(qbuf, kbuf, vbuf, bbuf, ybuf);

    // h = x + y @ Wo + bo (fused residual + bias epilogue)
    gemm_f16_wmma_kernel<<<ggrid, 256, 0, stream>>>(ybuf, Wot, hbuf, x, bo, MTOK, D_, D_);

    ln_kernel<<<MTOK, 256, 0, stream>>>(hbuf, lng, lnb, out);
}
